// PointConv_64269890617415
// MI455X (gfx1250) — matrix operand order repaired and checked by reading, not compiled, not run on hardware
//
#include <hip/hip_runtime.h>

typedef __attribute__((ext_vector_type(16))) _Float16 v16h;
typedef __attribute__((ext_vector_type(8)))  float    v8f;

#define NPTS 32768
#define KMAX 64
#define CIN  16
#define CMID 64
#define COUT 64
#define MT_STRIDE 72        // halves; 36 dwords -> conflict-free LDS row stepping
#define KTOT (CIN * CMID)   // 1024 = GEMM2 reduction length
#define NKC  (KTOT / 32)    // 32 k-chunks

__device__ __forceinline__ float celu1(float x) {
    return x > 0.0f ? x : expm1f(x);
}

// f16 A-matrix 16x32 layout: per (lane, vgpr) -> K offset (ISA 7.12.2)
__device__ __forceinline__ int a16_koff(int lane, int v) {
    int g = (lane >> 4) & 1;
    return (v < 4 ? 2 * v : 16 + 2 * (v - 4)) + 8 * g;
}
// f16 B-matrix 32x16 layout: per (lane, vgpr) -> K offset
__device__ __forceinline__ int b16_koff(int lane, int v) {
    int g = (lane >> 4) & 1;
    return (v < 4 ? 2 * v : 16 + 2 * (v - 4)) + 8 * g;
}
// inverse of a16_koff: K offset within 32-chunk -> (lane group, vgpr)
__device__ __forceinline__ void a16_inv(int koff, int& g, int& v) {
    g = (koff >> 3) & 1;          // +8g bands: 8..15 / 24..31
    int base = koff & ~8;         // 0..7 or 16..23
    v = base < 8 ? (base >> 1) : 4 + ((base - 16) >> 1);
}

// ---------------------------------------------------------------------------
// Kernel C: pack W3[1024x64] f32 -> f16 WMMA-B operand image (128 KB, L2-hot).
// W3op[kc][tile][lane][v] (uint = 2 halves): lane's 8 B-VGPRs contiguous.
__global__ __launch_bounds__(128)
void pc_pack_w3(const float* __restrict__ W3, unsigned* __restrict__ W3op) {
    const int t    = blockIdx.x * 128 + threadIdx.x;   // 0..4095
    const int lane = t & 31;
    const int tile = (t >> 5) & 3;
    const int kc   = t >> 7;
    const int n    = tile * 16 + (lane & 15);
    union { unsigned u[8]; uint4 q[2]; } o;
#pragma unroll
    for (int v = 0; v < 8; ++v) {
        int k = kc * 32 + b16_koff(lane, v);
        union { _Float16 h[2]; unsigned u; } pk;
        pk.h[0] = (_Float16)W3[k * COUT + n];
        pk.h[1] = (_Float16)W3[(k + 1) * COUT + n];
        o.u[v] = pk.u;
    }
    uint4* dst = (uint4*)(W3op + (size_t)t * 8);
    dst[0] = o.q[0];
    dst[1] = o.q[1];
}

// ---------------------------------------------------------------------------
// Kernel A: one block per output point. Edge pos-MLP (f32 VALU) -> LDS f16
// tiles -> WMMA GEMM1: P[16x64] = X^T[16xK] @ M[Kx64]; scatter P (f16) into
// the GEMM2 A-operand image Pop[pb][kc][lane][v].
__global__ __launch_bounds__(128)
void pc_edge_gemm1(const float* __restrict__ x_in,
                   const float* __restrict__ pos_in,
                   const float* __restrict__ pos_out,
                   const int*   __restrict__ in_index,
                   const int*   __restrict__ out_index,
                   const float* __restrict__ W1,
                   const float* __restrict__ W2,
                   unsigned*    __restrict__ Pop,
                   int E) {
    __shared__ float    W2s[CIN][CMID];
    __shared__ _Float16 Mt[CMID][MT_STRIDE];   // M^T: [cmid][k]
    __shared__ _Float16 Xt[CIN][MT_STRIDE];    // X^T: [cin][k]
    __shared__ int      seg[2];

    const int p   = blockIdx.x;
    const int tid = threadIdx.x;

    for (int i = tid; i < CIN * CMID; i += 128) W2s[i >> 6][i & 63] = W2[i];

    if (tid == 0) {
        // out_index is sorted: contiguous segment per point via binary search
        int lo = 0, hi = E;
        while (lo < hi) { int mid = (lo + hi) >> 1; if (out_index[mid] <  p) lo = mid + 1; else hi = mid; }
        seg[0] = lo;
        int lo2 = lo; hi = E;
        while (lo2 < hi) { int mid = (lo2 + hi) >> 1; if (out_index[mid] <= p) lo2 = mid + 1; else hi = mid; }
        seg[1] = lo2;
    }
    __syncthreads();

    const int   start    = seg[0];
    const int   cnt_full = seg[1] - seg[0];
    const int   c        = cnt_full < KMAX ? cnt_full : KMAX;
    const float inv_cf   = cnt_full > 0 ? 1.0f / (float)cnt_full : 0.0f;

    const int  k     = tid & 63;     // edge slot
    const int  half  = tid >> 6;     // which half of the 64 C_MID outputs
    const bool valid = (k < c);

    float px = 0.f, py = 0.f, pz = 0.f;
    int   inn = 0;
    if (valid) {
        inn = in_index[start + k];
        px = pos_in[inn * 3 + 0] - pos_out[p * 3 + 0];
        py = pos_in[inn * 3 + 1] - pos_out[p * 3 + 1];
        pz = pos_in[inn * 3 + 2] - pos_out[p * 3 + 2];
    }
    // h = celu(pos @ W1[3,16]); zero pos -> celu(0)=0, matching padded slots
    float h[16];
#pragma unroll
    for (int i = 0; i < 16; ++i)
        h[i] = celu1(px * W1[i] + py * W1[16 + i] + pz * W1[32 + i]);

    const int jbase = half * 32;
#pragma unroll
    for (int j = 0; j < 32; ++j) {
        float s = 0.f;
#pragma unroll
        for (int i = 0; i < 16; ++i) s += h[i] * W2s[i][jbase + j];
        Mt[jbase + j][k] = (_Float16)celu1(s);
    }
    if (half == 0) {
#pragma unroll
        for (int i = 0; i < 16; ++i) {
            float xv = valid ? x_in[inn * CIN + i] * inv_cf : 0.0f;
            Xt[i][k] = (_Float16)xv;
        }
    }
    __syncthreads();

    // GEMM1: 4 waves, one 16x16 N-tile each; K=64 -> 2 WMMA k-steps
    const int wave = tid >> 5;
    const int lane = tid & 31;
    const int row  = lane & 15;
    const int g    = (lane >> 4) & 1;

    v8f acc = {0.f, 0.f, 0.f, 0.f, 0.f, 0.f, 0.f, 0.f};
    const unsigned* XtU = (const unsigned*)&Xt[0][0];
    const unsigned* MtU = (const unsigned*)&Mt[0][0];
#pragma unroll
    for (int kk = 0; kk < KMAX; kk += 32) {
        union { unsigned u[8]; v16h v; } a, b;
#pragma unroll
        for (int v = 0; v < 8; ++v) {
            int ka = kk + a16_koff(lane, v);
            a.u[v] = XtU[(row * MT_STRIDE + ka) >> 1];                 // X^T[row][ka..ka+1]
            int kb = kk + b16_koff(lane, v);
            b.u[v] = MtU[((wave * 16 + row) * MT_STRIDE + kb) >> 1];   // M[kb][col] = M^T[col][kb]
        }
        acc = __builtin_amdgcn_wmma_f32_16x16x32_f16(false, a.v, false, b.v,
                                                     (short)0, acc, false, false);
    }

    // Scatter P into the GEMM2 A-operand image.
    // Element (crow=i+8g, mcol=wave*16+row): flat k = crow*64 + mcol, so
    // koff = mcol%32 (lane-constant), kc = 2*crow + wave/2 = kc0 + 2i.
    // => one base address, 8 stores at constant byte offsets (i * 2048 B).
    const int m    = wave * 16 + row;      // C_MID index
    const int koff = m & 31;
    int gB, vB; a16_inv(koff, gB, vB);
    const int laneB = (p & 15) + 16 * gB;  // point row within its 16-pt block
    const int pb    = p >> 4;
    const int kc0   = 16 * g + (wave >> 1);
    _Float16* base = (_Float16*)Pop
        + ((((size_t)pb * NKC + kc0) * 32 + laneB) * 8 + vB) * 2 + (koff & 1);
#pragma unroll
    for (int i = 0; i < 8; ++i)
        base[i * 1024] = (_Float16)acc[i];   // stride: 2 kc-slabs = 2048 bytes
}

// ---------------------------------------------------------------------------
// Kernel B: 16 points per block -> out[16x64] = P[16x1024] @ W3[1024x64] + b3.
// Operands pre-packed in WMMA register layout: 4x global_load_b128 + 1 WMMA
// per k-step; strided pointer bumps so unrolled deltas fold into load offsets.
__global__ __launch_bounds__(128)
void pc_gemm2(const unsigned* __restrict__ Pop,
              const unsigned* __restrict__ W3op,
              const float*    __restrict__ b3,
              float*          __restrict__ out) {
    const int pb   = blockIdx.x;
    const int tid  = threadIdx.x;
    const int wave = tid >> 5;       // N-tile (cols 16*wave ..)
    const int lane = tid & 31;
    const int row  = lane & 15;
    const int g    = (lane >> 4) & 1;
    const int ncol = wave * 16 + row;

    const uint4* aptr = (const uint4*)Pop  + ((size_t)pb * NKC * 32 + lane) * 2;
    const uint4* bptr = (const uint4*)W3op + ((size_t)wave * 32 + lane) * 2;

    v8f acc = {0.f, 0.f, 0.f, 0.f, 0.f, 0.f, 0.f, 0.f};
#pragma unroll 4
    for (int kc = 0; kc < NKC; ++kc) {
        union { uint4 q[2]; v16h v; } a, b;
        a.q[0] = aptr[0];
        a.q[1] = aptr[1];
        b.q[0] = bptr[0];
        b.q[1] = bptr[1];
        acc = __builtin_amdgcn_wmma_f32_16x16x32_f16(false, a.v, false, b.v,
                                                     (short)0, acc, false, false);
        aptr += 64;    // 32 lanes * 2 uint4 per kc-slab
        bptr += 256;   // 4 tiles * 32 lanes * 2 uint4 per kc-slab
    }

    const float bias = b3[ncol];
    float* obase = out + ((size_t)pb * 16 + 8 * g) * COUT + ncol;
#pragma unroll
    for (int i = 0; i < 8; ++i)
        obase[i * COUT] = acc[i] + bias;     // stride: 256 bytes, const offsets
}

// ---------------------------------------------------------------------------
extern "C" void kernel_launch(void* const* d_in, const int* in_sizes, int n_in,
                              void* d_out, int out_size, void* d_ws, size_t ws_size,
                              hipStream_t stream) {
    const float* x_in      = (const float*)d_in[0];
    const float* pos_in    = (const float*)d_in[1];
    const float* pos_out   = (const float*)d_in[2];
    const int*   in_index  = (const int*)d_in[3];
    const int*   out_index = (const int*)d_in[4];
    const float* W1        = (const float*)d_in[5];
    const float* W2        = (const float*)d_in[6];
    const float* W3        = (const float*)d_in[7];
    const float* b3        = (const float*)d_in[8];
    const int    E         = in_sizes[4];

    // ws layout: [0, 64MB) = P operand image (f16), [64MB, +128KB) = W3 image
    unsigned* Pop  = (unsigned*)d_ws;
    unsigned* W3op = (unsigned*)((char*)d_ws + (size_t)NPTS * KTOT * sizeof(_Float16));

    pc_pack_w3<<<32, 128, 0, stream>>>(W3, W3op);
    pc_edge_gemm1<<<NPTS, 128, 0, stream>>>(x_in, pos_in, pos_out,
                                            in_index, out_index, W1, W2, Pop, E);
    pc_gemm2<<<NPTS / 16, 128, 0, stream>>>(Pop, W3op, b3, (float*)d_out);
}
